// EncoderGraphGRU_35064113004950
// MI455X (gfx1250) — matrix operand order repaired and checked by reading, not compiled, not run on hardware
//
#include <hip/hip_runtime.h>
#include <math.h>

typedef unsigned short u16;
typedef __attribute__((ext_vector_type(16))) __bf16 v16bf;
typedef __attribute__((ext_vector_type(8)))  float  v8f;

struct __align__(16) U128 { unsigned x, y, z, w; };
struct B256 { U128 lo, hi; };

__device__ __forceinline__ u16 f2bf(float x) {
  unsigned u = __float_as_uint(x);
  u += 0x7FFFu + ((u >> 16) & 1u);          // round-to-nearest-even
  return (u16)(u >> 16);
}
__device__ __forceinline__ v8f zero8() { v8f v; for (int i = 0; i < 8; ++i) v[i] = 0.f; return v; }
__device__ __forceinline__ float sigm(float x) { return 1.0f / (1.0f + expf(-x)); }

// A fragment (16x32 bf16): lane<16 holds K {kb*32+0..7, +16..23}; lane>=16 holds {+8..15, +24..31}
__device__ __forceinline__ v16bf fragA(const u16* row, int kb, int hi) {
  const u16* p = row + kb * 32 + hi * 8;
  B256 t; t.lo = *(const U128*)p; t.hi = *(const U128*)(p + 16);
  return __builtin_bit_cast(v16bf, t);
}
// B fragment (32x16 bf16) from row-major W[N][K]: lane n holds W[n][kb*32 + {hi*8..+7, 16+hi*8..+7}]
__device__ __forceinline__ v16bf fragB(const u16* w, int K, int r, int kb, int hi) {
  const u16* p = w + (size_t)r * K + kb * 32 + hi * 8;
  B256 t; t.lo = *(const U128*)p; t.hi = *(const U128*)(p + 16);
  return __builtin_bit_cast(v16bf, t);
}
#define WMMA(a, b, c) __builtin_amdgcn_wmma_f32_16x16x32_bf16(false, (a), false, (b), (short)0, (c), false, false)

// ---------------- prep kernels ----------------
__global__ void k_fill0(float* p, long n) {
  long i = (long)blockIdx.x * blockDim.x + threadIdx.x, s = (long)gridDim.x * blockDim.x;
  for (; i < n; i += s) p[i] = 0.0f;
}
__global__ void k_cvt(const float* __restrict__ src, u16* __restrict__ dst, long n) {
  long i = (long)blockIdx.x * blockDim.x + threadIdx.x, s = (long)gridDim.x * blockDim.x;
  for (; i < n; i += s) dst[i] = f2bf(src[i]);
}
// dst[r][0:ka) = a[r], dst[r][ka:ka+kb) = b[r]   (fused wih|whh, bf16)
__global__ void k_cat2(const float* __restrict__ a, int ka, const float* __restrict__ b, int kb,
                       u16* __restrict__ dst, int rows) {
  long K = ka + kb, tot = (long)rows * K;
  long i = (long)blockIdx.x * blockDim.x + threadIdx.x, s = (long)gridDim.x * blockDim.x;
  for (; i < tot; i += s) {
    int r = (int)(i / K), c = (int)(i - (long)r * K);
    float v = (c < ka) ? a[(long)r * ka + c] : b[(long)r * kb + (c - ka)];
    dst[i] = f2bf(v);
  }
}

// ---------------- edge MLP + segment-max ----------------
// 32 edges per block; each wave computes a 32x64 output tile as two 16-row A tiles
// sharing every B fragment (8 WMMAs per 8 b128 B-loads per k-chunk).
__global__ __launch_bounds__(128) void edge_mlp_kernel(
    const float* __restrict__ x, const int* __restrict__ src, const int* __restrict__ dst,
    const u16* __restrict__ W1, const float* __restrict__ b1,
    const u16* __restrict__ W2, const float* __restrict__ b2,
    float* __restrict__ agg, int E) {
  __shared__ __align__(16) u16 INb[32][264];   // cat(x_i, x_j-x_i), K=256, +8 pad
  __shared__ __align__(16) u16 H1[32][264];
  __shared__ int sS[32], sD[32];
  int tid = threadIdx.x;
  int e0 = blockIdx.x * 32;
  if (tid < 32) { int e = e0 + tid; if (e >= E) e = E - 1; sS[tid] = src[e]; sD[tid] = dst[e]; }
  __syncthreads();
  for (int idx = tid; idx < 32 * 32; idx += 128) {
    int m = idx >> 5, c4 = idx & 31;
    const float4 xi = *(const float4*)(x + (size_t)sD[m] * 128 + c4 * 4);
    const float4 xj = *(const float4*)(x + (size_t)sS[m] * 128 + c4 * 4);
    int c = c4 * 4;
    INb[m][c + 0] = f2bf(xi.x); INb[m][c + 1] = f2bf(xi.y);
    INb[m][c + 2] = f2bf(xi.z); INb[m][c + 3] = f2bf(xi.w);
    INb[m][128 + c + 0] = f2bf(xj.x - xi.x); INb[m][128 + c + 1] = f2bf(xj.y - xi.y);
    INb[m][128 + c + 2] = f2bf(xj.z - xi.z); INb[m][128 + c + 3] = f2bf(xj.w - xi.w);
  }
  __syncthreads();
  int lane = tid & 31, wv = tid >> 5, hi = lane >> 4, ln = lane & 15;
  int c0 = wv * 64;
  v8f acc[2][4];
  for (int g = 0; g < 2; ++g) for (int t = 0; t < 4; ++t) acc[g][t] = zero8();
  for (int kb = 0; kb < 8; ++kb) {
    v16bf a0 = fragA(&INb[ln][0], kb, hi);
    v16bf a1 = fragA(&INb[16 + ln][0], kb, hi);
#pragma unroll
    for (int t = 0; t < 4; ++t) {
      v16bf b = fragB(W1, 256, c0 + t * 16 + ln, kb, hi);
      acc[0][t] = WMMA(a0, b, acc[0][t]);
      acc[1][t] = WMMA(a1, b, acc[1][t]);
    }
  }
#pragma unroll
  for (int g = 0; g < 2; ++g)
    for (int t = 0; t < 4; ++t)
      for (int r = 0; r < 8; ++r) {
        int n = c0 + t * 16 + ln, m = g * 16 + r + 8 * hi;
        H1[m][n] = f2bf(fmaxf(acc[g][t][r] + b1[n], 0.0f));
      }
  __syncthreads();
  for (int g = 0; g < 2; ++g) for (int t = 0; t < 4; ++t) acc[g][t] = zero8();
  for (int kb = 0; kb < 8; ++kb) {
    v16bf a0 = fragA(&H1[ln][0], kb, hi);
    v16bf a1 = fragA(&H1[16 + ln][0], kb, hi);
#pragma unroll
    for (int t = 0; t < 4; ++t) {
      v16bf b = fragB(W2, 256, c0 + t * 16 + ln, kb, hi);
      acc[0][t] = WMMA(a0, b, acc[0][t]);
      acc[1][t] = WMMA(a1, b, acc[1][t]);
    }
  }
#pragma unroll
  for (int g = 0; g < 2; ++g)
    for (int t = 0; t < 4; ++t)
      for (int r = 0; r < 8; ++r) {
        int n = c0 + t * 16 + ln, m = g * 16 + r + 8 * hi;
        float v = fmaxf(acc[g][t][r] + b2[n], 0.0f);   // relu >= 0, so agg init 0 == segment_max + neginf->0
        atomicMax(&agg[(size_t)sD[m] * 256 + n], v);   // global_atomic_max_num_f32
      }
}

// ---------------- node pipeline: hi-init + 2-layer GRU + VAE head ----------------
__global__ __launch_bounds__(128) void node_kernel(
    const float* __restrict__ agg, const float* __restrict__ tin, const float* __restrict__ ain,
    const float* __restrict__ eps,
    const u16* __restrict__ Hi0, const float* __restrict__ hb0,
    const u16* __restrict__ Hi1, const float* __restrict__ hb1,
    const u16* __restrict__ Wg0, const float* __restrict__ bih0, const float* __restrict__ bhh0,
    const u16* __restrict__ Wg1, const float* __restrict__ bih1, const float* __restrict__ bhh1,
    const u16* __restrict__ Mu, const float* __restrict__ mub,
    const u16* __restrict__ Lv, const float* __restrict__ lvb,
    float* __restrict__ out, int N) {
  __shared__ __align__(16) u16 FEAT[16][552];  // [feat(288) | h0(256)] bf16, +8 pad
  __shared__ __align__(16) u16 NH[16][520];    // [new_h0(256) | h1(256)] bf16, +8 pad
  __shared__ __align__(16) u16 NH1[16][264];   // new_h1 bf16
  __shared__ float H1F[16][256];               // h1 f32 for gate combine
  int tid = threadIdx.x;
  int n0 = blockIdx.x * 16;
  // P1: feat = [agg | t | a] -> bf16 LDS
  for (int idx = tid; idx < 16 * 288; idx += 128) {
    int m = idx / 288, c = idx - m * 288;
    int node = n0 + m; if (node >= N) node = N - 1;
    float v;
    if (c < 256)      v = agg[(size_t)node * 256 + c];
    else if (c < 272) v = tin[(size_t)node * 16 + (c - 256)];
    else              v = ain[(size_t)node * 16 + (c - 272)];
    FEAT[m][c] = f2bf(v);
  }
  __syncthreads();
  int lane = tid & 31, wv = tid >> 5, hi = lane >> 4, ln = lane & 15;
  int c0 = wv * 64;
  // P2: h1 = tanh(feat @ hi_w1^T + b)
  {
    v8f acc[4]; for (int t = 0; t < 4; ++t) acc[t] = zero8();
    for (int kb = 0; kb < 9; ++kb) {
      v16bf a = fragA(&FEAT[ln][0], kb, hi);
#pragma unroll
      for (int t = 0; t < 4; ++t)
        acc[t] = WMMA(a, fragB(Hi1, 288, c0 + t * 16 + ln, kb, hi), acc[t]);
    }
#pragma unroll
    for (int t = 0; t < 4; ++t)
      for (int r = 0; r < 8; ++r) {
        int n = c0 + t * 16 + ln, m = r + 8 * hi;
        float v = tanhf(acc[t][r] + hb1[n]);
        H1F[m][n] = v;
        NH[m][256 + n] = f2bf(v);
      }
  }
  // P3: h0 (kept in registers for the gate combine), bf16 copy appended to FEAT
  v8f h0v[4]; for (int t = 0; t < 4; ++t) h0v[t] = zero8();
  for (int kb = 0; kb < 9; ++kb) {
    v16bf a = fragA(&FEAT[ln][0], kb, hi);
#pragma unroll
    for (int t = 0; t < 4; ++t)
      h0v[t] = WMMA(a, fragB(Hi0, 288, c0 + t * 16 + ln, kb, hi), h0v[t]);
  }
#pragma unroll
  for (int t = 0; t < 4; ++t)
    for (int r = 0; r < 8; ++r) {
      int n = c0 + t * 16 + ln, m = r + 8 * hi;
      float v = tanhf(h0v[t][r] + hb0[n]);
      h0v[t][r] = v;
      FEAT[m][288 + n] = f2bf(v);
    }
  __syncthreads();
  // P4: GRU layer 0 on fused A=[feat|h0] (K=544) and Wg0=[wih|whh] (768x544)
  v8f ar[4], az[4], ai[4], ah[4];
  for (int t = 0; t < 4; ++t) { ar[t] = zero8(); az[t] = zero8(); ai[t] = zero8(); ah[t] = zero8(); }
  for (int kb = 0; kb < 17; ++kb) {
    v16bf a = fragA(&FEAT[ln][0], kb, hi);
#pragma unroll
    for (int t = 0; t < 4; ++t) {
      int n = c0 + t * 16 + ln;
      ar[t] = WMMA(a, fragB(Wg0, 544, n, kb, hi), ar[t]);
      az[t] = WMMA(a, fragB(Wg0, 544, 256 + n, kb, hi), az[t]);
      v16bf bn = fragB(Wg0, 544, 512 + n, kb, hi);
      if (kb < 9) ai[t] = WMMA(a, bn, ai[t]);   // i_n over feat (K 0..287)
      else        ah[t] = WMMA(a, bn, ah[t]);   // h_n over h0   (K 288..543)
    }
  }
#pragma unroll
  for (int t = 0; t < 4; ++t)
    for (int r = 0; r < 8; ++r) {
      int n = c0 + t * 16 + ln, m = r + 8 * hi, node = n0 + m;
      float rg = sigm(ar[t][r] + bih0[n] + bhh0[n]);
      float zg = sigm(az[t][r] + bih0[256 + n] + bhh0[256 + n]);
      float ng = tanhf(ai[t][r] + bih0[512 + n] + rg * (ah[t][r] + bhh0[512 + n]));
      float nh = (1.0f - zg) * ng + zg * h0v[t][r];
      if (node < N) out[(size_t)node * 256 + n] = nh;      // h[0]
      NH[m][n] = f2bf(nh);
    }
  __syncthreads();
  // P5: h1 -> registers (same accumulator layout)
  v8f h1v[4];
#pragma unroll
  for (int t = 0; t < 4; ++t)
    for (int r = 0; r < 8; ++r)
      h1v[t][r] = H1F[r + 8 * hi][c0 + t * 16 + ln];
  // P6: GRU layer 1 on A=[new_h0|h1] (K=512), Wg1 (768x512)
  for (int t = 0; t < 4; ++t) { ar[t] = zero8(); az[t] = zero8(); ai[t] = zero8(); ah[t] = zero8(); }
  for (int kb = 0; kb < 16; ++kb) {
    v16bf a = fragA(&NH[ln][0], kb, hi);
#pragma unroll
    for (int t = 0; t < 4; ++t) {
      int n = c0 + t * 16 + ln;
      ar[t] = WMMA(a, fragB(Wg1, 512, n, kb, hi), ar[t]);
      az[t] = WMMA(a, fragB(Wg1, 512, 256 + n, kb, hi), az[t]);
      v16bf bn = fragB(Wg1, 512, 512 + n, kb, hi);
      if (kb < 8) ai[t] = WMMA(a, bn, ai[t]);
      else        ah[t] = WMMA(a, bn, ah[t]);
    }
  }
#pragma unroll
  for (int t = 0; t < 4; ++t)
    for (int r = 0; r < 8; ++r) {
      int n = c0 + t * 16 + ln, m = r + 8 * hi, node = n0 + m;
      float rg = sigm(ar[t][r] + bih1[n] + bhh1[n]);
      float zg = sigm(az[t][r] + bih1[256 + n] + bhh1[256 + n]);
      float ng = tanhf(ai[t][r] + bih1[512 + n] + rg * (ah[t][r] + bhh1[512 + n]));
      float nh = (1.0f - zg) * ng + zg * h1v[t][r];
      if (node < N) out[(size_t)N * 256 + (size_t)node * 256 + n] = nh;  // h[1]
      NH1[m][n] = f2bf(nh);
    }
  __syncthreads();
  // P7: mu / log_var / z  (wave wv owns col tile wv of 16)
  {
    v8f amu = zero8(), alv = zero8();
    int n = wv * 16 + ln;
    for (int kb = 0; kb < 8; ++kb) {
      v16bf a = fragA(&NH1[ln][0], kb, hi);
      amu = WMMA(a, fragB(Mu, 256, n, kb, hi), amu);
      alv = WMMA(a, fragB(Lv, 256, n, kb, hi), alv);
    }
    size_t MU = (size_t)2 * N * 256;
    size_t LVo = MU + (size_t)N * 64;
    size_t ZO = LVo + (size_t)N * 64;
#pragma unroll
    for (int r = 0; r < 8; ++r) {
      int m = r + 8 * hi, node = n0 + m;
      if (node >= N) continue;
      float mu = amu[r] + mub[n];
      float lv = alv[r] + lvb[n];
      float zz = mu + eps[(size_t)node * 64 + n] * expf(0.5f * lv);
      out[MU + (size_t)node * 64 + n] = mu;
      out[LVo + (size_t)node * 64 + n] = lv;
      out[ZO + (size_t)node * 64 + n] = zz;
    }
  }
}

extern "C" void kernel_launch(void* const* d_in, const int* in_sizes, int n_in,
                              void* d_out, int out_size, void* d_ws, size_t ws_size,
                              hipStream_t stream) {
  const float* x      = (const float*)d_in[0];
  const float* t      = (const float*)d_in[1];
  const float* a      = (const float*)d_in[2];
  const float* eps    = (const float*)d_in[3];
  const int*   ei     = (const int*)d_in[4];
  const float* enc_w1 = (const float*)d_in[5];
  const float* enc_b1 = (const float*)d_in[6];
  const float* enc_w2 = (const float*)d_in[7];
  const float* enc_b2 = (const float*)d_in[8];
  const float* hi_w0  = (const float*)d_in[9];
  const float* hi_b0  = (const float*)d_in[10];
  const float* hi_w1  = (const float*)d_in[11];
  const float* hi_b1  = (const float*)d_in[12];
  const float* wih0   = (const float*)d_in[13];
  const float* whh0   = (const float*)d_in[14];
  const float* bih0   = (const float*)d_in[15];
  const float* bhh0   = (const float*)d_in[16];
  const float* wih1   = (const float*)d_in[17];
  const float* whh1   = (const float*)d_in[18];
  const float* bih1   = (const float*)d_in[19];
  const float* bhh1   = (const float*)d_in[20];
  const float* mu_w   = (const float*)d_in[21];
  const float* mu_b   = (const float*)d_in[22];
  const float* lv_w   = (const float*)d_in[23];
  const float* lv_b   = (const float*)d_in[24];
  int N = in_sizes[0] / 128;
  int E = in_sizes[4] / 2;
  const int* srcIdx = ei;
  const int* dstIdx = ei + E;

  char* ws = (char*)d_ws;
  size_t off = 0;
  auto take = [&](size_t b) { void* p = ws + off; off += (b + 255) & ~(size_t)255; return p; };
  float* agg = (float*)take((size_t)N * 256 * 4);
  u16* W1b  = (u16*)take(256 * 256 * 2);
  u16* W2b  = (u16*)take(256 * 256 * 2);
  u16* Hi0b = (u16*)take(256 * 288 * 2);
  u16* Hi1b = (u16*)take(256 * 288 * 2);
  u16* Wg0b = (u16*)take(768 * 544 * 2);
  u16* Wg1b = (u16*)take(768 * 512 * 2);
  u16* Mub  = (u16*)take(64 * 256 * 2);
  u16* Lvb  = (u16*)take(64 * 256 * 2);

  k_fill0<<<dim3(2048), dim3(256), 0, stream>>>(agg, (long)N * 256);
  k_cvt<<<dim3(128), dim3(256), 0, stream>>>(enc_w1, W1b, 256 * 256);
  k_cvt<<<dim3(128), dim3(256), 0, stream>>>(enc_w2, W2b, 256 * 256);
  k_cvt<<<dim3(128), dim3(256), 0, stream>>>(hi_w0, Hi0b, 256 * 288);
  k_cvt<<<dim3(128), dim3(256), 0, stream>>>(hi_w1, Hi1b, 256 * 288);
  k_cvt<<<dim3(64),  dim3(256), 0, stream>>>(mu_w, Mub, 64 * 256);
  k_cvt<<<dim3(64),  dim3(256), 0, stream>>>(lv_w, Lvb, 64 * 256);
  k_cat2<<<dim3(512), dim3(256), 0, stream>>>(wih0, 288, whh0, 256, Wg0b, 768);
  k_cat2<<<dim3(512), dim3(256), 0, stream>>>(wih1, 256, whh1, 256, Wg1b, 768);

  edge_mlp_kernel<<<dim3((E + 31) / 32), dim3(128), 0, stream>>>(
      x, srcIdx, dstIdx, W1b, enc_b1, W2b, enc_b2, agg, E);
  node_kernel<<<dim3((N + 15) / 16), dim3(128), 0, stream>>>(
      agg, t, a, eps, Hi0b, hi_b0, Hi1b, hi_b1,
      Wg0b, bih0, bhh0, Wg1b, bih1, bhh1,
      Mub, mu_b, Lvb, lv_b, (float*)d_out, N);
}
